// CausalSelfAttention_59253368815644
// MI455X (gfx1250) — matrix operand order repaired and checked by reading, not compiled, not run on hardware
//
#include <hip/hip_runtime.h>
#include <hip/hip_bf16.h>
#include <math.h>

// Shapes fixed by the reference.
#define Bz   2
#define Tz   2048
#define Dz   1024
#define Hz   16
#define HDz  64
#define E3z  (3 * Dz)
#define EPSz 1.1920929e-07f

typedef __attribute__((ext_vector_type(16))) _Float16 v16h;
typedef __attribute__((ext_vector_type(8)))  _Float16 v8h;
typedef __attribute__((ext_vector_type(8)))  float    v8f;

union ABu { v16h v; v8h h[2]; };

static __device__ inline float red_sum32(float v) {
    v += __shfl_xor(v, 16, 32);
    v += __shfl_xor(v, 8, 32);
    v += __shfl_xor(v, 4, 32);
    v += __shfl_xor(v, 2, 32);
    v += __shfl_xor(v, 1, 32);
    return v;
}
static __device__ inline float red_max16(float v) {
    v = fmaxf(v, __shfl_xor(v, 1, 16));
    v = fmaxf(v, __shfl_xor(v, 2, 16));
    v = fmaxf(v, __shfl_xor(v, 4, 16));
    v = fmaxf(v, __shfl_xor(v, 8, 16));
    return v;
}
static __device__ inline float red_sum16(float v) {
    v += __shfl_xor(v, 1, 16);
    v += __shfl_xor(v, 2, 16);
    v += __shfl_xor(v, 4, 16);
    v += __shfl_xor(v, 8, 16);
    return v;
}

// 16-byte async global->LDS copy (per lane), tracked by ASYNCcnt.
static __device__ inline void async_b128(unsigned lds_off, const void* gptr) {
    unsigned long long ga = (unsigned long long)(size_t)gptr;
    asm volatile("global_load_async_to_lds_b128 %0, %1, off"
                 :: "v"(lds_off), "v"(ga) : "memory");
}

// ---------------- conversions ----------------

// wh[e*D+d] = w[e*D+d] * (e < 3D ? lam0 : lam1)
__global__ __launch_bounds__(256) void k_convert_w(const float* __restrict__ w,
                                                   const float* __restrict__ lam,
                                                   _Float16* __restrict__ wh) {
    int i = blockIdx.x * 256 + threadIdx.x;           // i < 4D*D = 4194304
    int e = i >> 10;
    float l = (e < E3z) ? lam[0] : lam[1];
    wh[i] = (_Float16)(w[i] * l);
}

__global__ __launch_bounds__(256) void k_convert_x(const float* __restrict__ x,
                                                   _Float16* __restrict__ xh) {
    int i = blockIdx.x * 256 + threadIdx.x;           // i < B*T*D = 4194304
    xh[i] = (_Float16)x[i];
}

// ---------------- generic f16 WMMA GEMM: C[m,n] = sum_k A[m,k]*Bw[n,k] ----------------
// One wave computes a 16(M) x 64(N) tile. A: MxK row-major, Bw: NxK row-major, C: MxN f32.
__global__ __launch_bounds__(256) void k_gemm16(const _Float16* __restrict__ A,
                                                const _Float16* __restrict__ Bw,
                                                float* __restrict__ C,
                                                int Mt, int Nt, int K, int N) {
    int wid  = threadIdx.x >> 5;
    int lane = threadIdx.x & 31;
    int job  = blockIdx.x * (blockDim.x >> 5) + wid;
    if (job >= Mt * Nt) return;
    int mt = job / Nt;
    int nt = job % Nt;

    int lane16 = lane & 15;
    int hf     = lane >> 4;          // lane half
    int koff   = hf ? 16 : 0;        // A-fragment K sub-offset
    int khalf  = hf ? 16 : 0;        // B-fragment K sub-offset

    const _Float16* arow = A + (size_t)(mt * 16 + lane16) * K;
    const _Float16* b0   = Bw + (size_t)(nt * 64 + 0 * 16 + lane16) * K + khalf;
    const _Float16* b1   = Bw + (size_t)(nt * 64 + 1 * 16 + lane16) * K + khalf;
    const _Float16* b2   = Bw + (size_t)(nt * 64 + 2 * 16 + lane16) * K + khalf;
    const _Float16* b3   = Bw + (size_t)(nt * 64 + 3 * 16 + lane16) * K + khalf;

    v8f c0 = {}, c1 = {}, c2 = {}, c3 = {};

    for (int k0 = 0; k0 < K; k0 += 32) {
        __builtin_prefetch(arow + k0 + 256, 0, 1);     // global_prefetch_b8
        __builtin_prefetch(b0 + k0 + 256, 0, 1);

        ABu a;
        a.h[0] = *(const v8h*)(arow + k0 + koff);      // K = koff..koff+7
        a.h[1] = *(const v8h*)(arow + k0 + koff + 8);  // K = koff+8..koff+15
        ABu f0, f1, f2, f3;
        f0.h[0] = *(const v8h*)(b0 + k0);     f0.h[1] = *(const v8h*)(b0 + k0 + 8);
        f1.h[0] = *(const v8h*)(b1 + k0);     f1.h[1] = *(const v8h*)(b1 + k0 + 8);
        f2.h[0] = *(const v8h*)(b2 + k0);     f2.h[1] = *(const v8h*)(b2 + k0 + 8);
        f3.h[0] = *(const v8h*)(b3 + k0);     f3.h[1] = *(const v8h*)(b3 + k0 + 8);

        c0 = __builtin_amdgcn_wmma_f32_16x16x32_f16(false, a.v, false, f0.v, (short)0, c0, false, false);
        c1 = __builtin_amdgcn_wmma_f32_16x16x32_f16(false, a.v, false, f1.v, (short)0, c1, false, false);
        c2 = __builtin_amdgcn_wmma_f32_16x16x32_f16(false, a.v, false, f2.v, (short)0, c2, false, false);
        c3 = __builtin_amdgcn_wmma_f32_16x16x32_f16(false, a.v, false, f3.v, (short)0, c3, false, false);
    }

    int m0 = mt * 16, n0 = nt * 64;
    int rr = hf ? 8 : 0;
#pragma unroll
    for (int r = 0; r < 8; ++r) {
        float* crow = C + (size_t)(m0 + r + rr) * N + n0 + lane16;
        crow[0]  = c0[r];
        crow[16] = c1[r];
        crow[32] = c2[r];
        crow[48] = c3[r];
    }
}

// ---------------- gate GEMVs: sigmoid(x . w_h) ----------------
__global__ __launch_bounds__(256) void k_gates(const float* __restrict__ x,
                                               const float* __restrict__ wa,
                                               const float* __restrict__ wv,
                                               float* __restrict__ ga,
                                               float* __restrict__ gv) {
    int wid  = threadIdx.x >> 5;
    int lane = threadIdx.x & 31;
    int job  = blockIdx.x * 8 + wid;           // 65536 = B*T*H
    int h    = job & 15;
    int bt   = job >> 4;
    const float* xr  = x  + (size_t)bt * Dz;
    const float* war = wa + (size_t)h * Dz;
    const float* wvr = wv + (size_t)h * Dz;
    float sa = 0.f, sv = 0.f;
    for (int d = lane; d < Dz; d += 32) {
        float xv = xr[d];
        sa += xv * war[d];
        sv += xv * wvr[d];
    }
    sa = red_sum32(sa);
    sv = red_sum32(sv);
    if (lane == 0) {
        ga[(size_t)bt * Hz + h] = 1.0f / (1.0f + __expf(-sa));
        gv[(size_t)bt * Hz + h] = 2.0f / (1.0f + __expf(-sv));
    }
}

// ---------------- RMS norm + rotary + key-offset + v gate ----------------
__global__ __launch_bounds__(256) void k_post(const float* __restrict__ qkv,
                                              const float* __restrict__ ve,
                                              const float* __restrict__ cosp,
                                              const float* __restrict__ sinp,
                                              const float* __restrict__ gv,
                                              _Float16* __restrict__ qh,
                                              _Float16* __restrict__ kh,
                                              _Float16* __restrict__ vT) {
    int wid  = threadIdx.x >> 5;
    int d    = threadIdx.x & 31;               // lane handles dims d and d+32
    int job  = blockIdx.x * 8 + wid;           // 65536 = B*T*H
    int h    = job & 15;
    int bt   = job >> 4;
    int b    = bt >> 11;
    int t    = bt & (Tz - 1);
    int bh   = b * Hz + h;

    const float* row = qkv + (size_t)bt * E3z + h * HDz;
    float q1 = row[d],            q2 = row[d + 32];
    float k1 = row[Dz + d],       k2 = row[Dz + d + 32];
    float v1 = row[2 * Dz + d],   v2 = row[2 * Dz + d + 32];

    float msq = red_sum32(q1 * q1 + q2 * q2) * (1.0f / HDz);
    float msk = red_sum32(k1 * k1 + k2 * k2) * (1.0f / HDz);
    float rq = rsqrtf(msq + EPSz);
    float rk = rsqrtf(msk + EPSz);
    q1 *= rq; q2 *= rq; k1 *= rk; k2 *= rk;

    float c = cosp[t * 32 + d], s = sinp[t * 32 + d];
    float qo1 =  q1 * c + q2 * s;
    float qo2 = -q1 * s + q2 * c;
    float ko1 =  k1 * c + k2 * s;
    float ko2 = -k1 * s + k2 * c;

    size_t qb = ((size_t)bh * Tz + t) * HDz;
    qh[qb + d]      = (_Float16)qo1;
    qh[qb + d + 32] = (_Float16)qo2;

    // KEY_OFFSET: dims [16,32) and [48,64) of row t+1 come from row t.
    if (d < 16) {
        kh[qb + d]      = (_Float16)ko1;
        kh[qb + d + 32] = (_Float16)ko2;
    } else {
        if (t + 1 < Tz) {
            size_t kb2 = qb + HDz;             // row t+1
            kh[kb2 + d]      = (_Float16)ko1;
            kh[kb2 + d + 32] = (_Float16)ko2;
        }
        if (t == 0) {                          // row 0 keeps its own
            kh[qb + d]      = (_Float16)ko1;
            kh[qb + d + 32] = (_Float16)ko2;
        }
    }

    float g   = gv[(size_t)bt * Hz + h];
    float ve1 = ve[(size_t)bt * Dz + h * HDz + d];
    float ve2 = ve[(size_t)bt * Dz + h * HDz + d + 32];
    v1 += g * ve1;
    v2 += g * ve2;
    size_t vb = ((size_t)bh * HDz + d) * Tz + t;   // transposed [b,h,hd,t]
    vT[vb]                    = (_Float16)v1;
    vT[vb + (size_t)32 * Tz]  = (_Float16)v2;
}

// ---------------- flash attention: one wave per 16-row q tile ----------------
// K/V tiles staged into LDS with GLOBAL_LOAD_ASYNC_TO_LDS_B128 (ASYNCcnt),
// fragments read back with ds_load_b128.
__global__ __launch_bounds__(256) void k_attn(const _Float16* __restrict__ qh,
                                              const _Float16* __restrict__ kh,
                                              const _Float16* __restrict__ vT,
                                              const float* __restrict__ ga,
                                              const int* __restrict__ docs,
                                              const float* __restrict__ scale_p,
                                              _Float16* __restrict__ yh) {
    __shared__ __align__(16) _Float16 kbuf[8][32 * 64];   // 32 KB: rows k0..k0+31 x 64 dims
    __shared__ __align__(16) _Float16 vbuf[8][64 * 32];   // 32 KB: 64 hd-rows x 32 keys
    __shared__ __align__(16) _Float16 pbuf[8][16 * 32];   //  8 KB
    __shared__ int sdocs[Tz];                             //  8 KB
    for (int i = threadIdx.x; i < Tz; i += 256) sdocs[i] = docs[i];
    __syncthreads();

    int wid  = threadIdx.x >> 5;
    int lane = threadIdx.x & 31;
    int job  = blockIdx.x * 8 + wid;          // 4096 = B*H*(T/16)
    int qt   = job & 127;
    int bh   = job >> 7;
    int b    = bh >> 4;
    int h    = bh & 15;
    int q0   = qt * 16;

    int lane16 = lane & 15;
    int hf     = lane >> 4;
    int koff   = hf ? 16 : 0;
    int khalf  = hf ? 16 : 0;

    // Q A-fragments (d 0..31 and 32..63)
    const _Float16* qrow = qh + ((size_t)bh * Tz + q0 + lane16) * HDz;
    ABu aq0, aq1;
    aq0.h[0] = *(const v8h*)(qrow + koff);
    aq0.h[1] = *(const v8h*)(qrow + koff + 8);
    aq1.h[0] = *(const v8h*)(qrow + 32 + koff);
    aq1.h[1] = *(const v8h*)(qrow + 32 + koff + 8);

    const _Float16* khb = kh + (size_t)bh * Tz * HDz;
    const _Float16* vtb = vT + (size_t)bh * HDz * Tz;

    unsigned lds_k = (unsigned)(size_t)&kbuf[wid][0];
    unsigned lds_v = (unsigned)(size_t)&vbuf[wid][0];
    const _Float16* ksh = &kbuf[wid][0];
    const _Float16* vsh = &vbuf[wid][0];

    v8f o0 = {}, o1 = {}, o2 = {}, o3 = {};
    float mi[8], li[8];
    int dq[8];
#pragma unroll
    for (int r = 0; r < 8; ++r) {
        mi[r] = -__builtin_inff();
        li[r] = 0.f;
        dq[r] = sdocs[q0 + r + 8 * hf];
    }
    float scale = scale_p[0];

    for (int k0 = 0; k0 < q0 + 16; k0 += 32) {
        // ---- async-stage K tile: contiguous 4 KB (32 rows x 64 halves) ----
        {
            const _Float16* gk = khb + (size_t)k0 * HDz;
#pragma unroll
            for (int j = 0; j < 8; ++j) {
                int c = j * 32 + lane;                     // 16B chunk index
                async_b128(lds_k + c * 16, gk + c * 8);
            }
        }
        // ---- async-stage V tile: 64 hd-rows x 32 keys (64 B per row) ----
        {
#pragma unroll
            for (int j = 0; j < 8; ++j) {
                int c    = j * 32 + lane;
                int rowd = c >> 2;                         // hd row 0..63
                int sub  = c & 3;                          // 16B chunk in row
                async_b128(lds_v + rowd * 64 + sub * 16,
                           vtb + (size_t)rowd * Tz + k0 + sub * 8);
            }
        }
        asm volatile("s_wait_asynccnt 0x0" ::: "memory");

        // ---- scores for two 16-key subtiles (K fragments from LDS) ----
        v8f s0, s1;
        {
            const _Float16* krow = ksh + (size_t)lane16 * HDz;
            ABu f0, f1;
            f0.h[0] = *(const v8h*)(krow + khalf);      f0.h[1] = *(const v8h*)(krow + khalf + 8);
            f1.h[0] = *(const v8h*)(krow + 32 + khalf); f1.h[1] = *(const v8h*)(krow + 32 + khalf + 8);
            v8f ss = {};
            ss = __builtin_amdgcn_wmma_f32_16x16x32_f16(false, aq0.v, false, f0.v, (short)0, ss, false, false);
            ss = __builtin_amdgcn_wmma_f32_16x16x32_f16(false, aq1.v, false, f1.v, (short)0, ss, false, false);
            s0 = ss;
        }
        {
            const _Float16* krow = ksh + (size_t)(16 + lane16) * HDz;
            ABu f0, f1;
            f0.h[0] = *(const v8h*)(krow + khalf);      f0.h[1] = *(const v8h*)(krow + khalf + 8);
            f1.h[0] = *(const v8h*)(krow + 32 + khalf); f1.h[1] = *(const v8h*)(krow + 32 + khalf + 8);
            v8f ss = {};
            ss = __builtin_amdgcn_wmma_f32_16x16x32_f16(false, aq0.v, false, f0.v, (short)0, ss, false, false);
            ss = __builtin_amdgcn_wmma_f32_16x16x32_f16(false, aq1.v, false, f1.v, (short)0, ss, false, false);
            s1 = ss;
        }

        int kidx0 = k0 + lane16;
        int kidx1 = k0 + 16 + lane16;
        int dk0 = sdocs[kidx0];
        int dk1 = sdocs[kidx1];

        // ---- mask + online softmax per row ----
#pragma unroll
        for (int r = 0; r < 8; ++r) {
            int q = q0 + r + 8 * hf;
            float sv0 = ((q >= kidx0) && (dq[r] == dk0)) ? s0[r] * scale : -__builtin_inff();
            float sv1 = ((q >= kidx1) && (dq[r] == dk1)) ? s1[r] * scale : -__builtin_inff();
            float rm = red_max16(fmaxf(sv0, sv1));
            float mn = fmaxf(mi[r], rm);
            float msafe = (mn == -__builtin_inff()) ? 0.f : mn;
            float p0 = __expf(sv0 - msafe);
            float p1 = __expf(sv1 - msafe);
            float alpha = __expf(mi[r] - msafe);
            li[r] = li[r] * alpha + red_sum16(p0 + p1);
            mi[r] = mn;
            o0[r] *= alpha; o1[r] *= alpha; o2[r] *= alpha; o3[r] *= alpha;
            int m = r + 8 * hf;
            pbuf[wid][m * 32 + lane16]      = (_Float16)p0;
            pbuf[wid][m * 32 + 16 + lane16] = (_Float16)p1;
        }

        asm volatile("s_wait_dscnt 0x0" ::: "memory");

        // ---- reload P as A-fragment (16 x 32) ----
        ABu pa;
        const _Float16* pr = &pbuf[wid][lane16 * 32];
        pa.h[0] = *(const v8h*)(pr + koff);
    pa.h[1] = *(const v8h*)(pr + koff + 8);

        // ---- P . V  (V fragments from LDS) ----
        {
            const _Float16* vrow = vsh + (size_t)(0 * 16 + lane16) * 32 + khalf;
            ABu bv; bv.h[0] = *(const v8h*)(vrow); bv.h[1] = *(const v8h*)(vrow + 8);
            o0 = __builtin_amdgcn_wmma_f32_16x16x32_f16(false, pa.v, false, bv.v, (short)0, o0, false, false);
        }
        {
            const _Float16* vrow = vsh + (size_t)(1 * 16 + lane16) * 32 + khalf;
            ABu bv; bv.h[0] = *(const v8h*)(vrow); bv.h[1] = *(const v8h*)(vrow + 8);
            o1 = __builtin_amdgcn_wmma_f32_16x16x32_f16(false, pa.v, false, bv.v, (short)0, o1, false, false);
        }
        {
            const _Float16* vrow = vsh + (size_t)(2 * 16 + lane16) * 32 + khalf;
            ABu bv; bv.h[0] = *(const v8h*)(vrow); bv.h[1] = *(const v8h*)(vrow + 8);
            o2 = __builtin_amdgcn_wmma_f32_16x16x32_f16(false, pa.v, false, bv.v, (short)0, o2, false, false);
        }
        {
            const _Float16* vrow = vsh + (size_t)(3 * 16 + lane16) * 32 + khalf;
            ABu bv; bv.h[0] = *(const v8h*)(vrow); bv.h[1] = *(const v8h*)(vrow + 8);
            o3 = __builtin_amdgcn_wmma_f32_16x16x32_f16(false, pa.v, false, bv.v, (short)0, o3, false, false);
        }
    }

    // ---- finalize: attn_gate / l, store f16 y ----
#pragma unroll
    for (int r = 0; r < 8; ++r) {
        int q = q0 + r + 8 * hf;
        float g = ga[((size_t)b * Tz + q) * Hz + h];
        float inv = g / li[r];
        _Float16* yrow = yh + ((size_t)b * Tz + q) * Dz + h * HDz + lane16;
        yrow[0]  = (_Float16)(o0[r] * inv);
        yrow[16] = (_Float16)(o1[r] * inv);
        yrow[32] = (_Float16)(o2[r] * inv);
        yrow[48] = (_Float16)(o3[r] * inv);
    }
}

// ---------------- launch ----------------
extern "C" void kernel_launch(void* const* d_in, const int* in_sizes, int n_in,
                              void* d_out, int out_size, void* d_ws, size_t ws_size,
                              hipStream_t stream) {
    const float* x     = (const float*)d_in[0];
    const float* ve    = (const float*)d_in[1];
    const float* lam   = (const float*)d_in[2];
    const float* cosp  = (const float*)d_in[3];
    const float* sinp  = (const float*)d_in[4];
    const float* qkvw  = (const float*)d_in[5];
    const float* agw   = (const float*)d_in[6];
    const float* vgw   = (const float*)d_in[7];
    const float* scale = (const float*)d_in[8];
    const int*   docs  = (const int*)d_in[9];
    float* out = (float*)d_out;

    char* w = (char*)d_ws;
    _Float16* xh   = (_Float16*)(w);                          //  8 MB
    _Float16* wh   = (_Float16*)(w + ((size_t)8  << 20));     //  8 MB
    float*    qkvf = (float*)   (w + ((size_t)16 << 20));     // 48 MB
    _Float16* qh   = (_Float16*)(w + ((size_t)64 << 20));     //  8 MB
    _Float16* kh   = (_Float16*)(w + ((size_t)72 << 20));     //  8 MB
    _Float16* vT   = (_Float16*)(w + ((size_t)80 << 20));     //  8 MB
    _Float16* yh   = (_Float16*)(w + ((size_t)88 << 20));     //  8 MB
    float*    ga   = (float*)   (w + ((size_t)96 << 20));     // 256 KB
    float*    gv   = (float*)   (w + ((size_t)96 << 20) + ((size_t)1 << 20));

    k_convert_w<<<16384, 256, 0, stream>>>(qkvw, lam, wh);
    k_convert_x<<<16384, 256, 0, stream>>>(x, xh);
    // qkv = xh . wh[0:3D]^T : M=4096 (256 tiles), N=3072 (48 tiles of 64), K=1024
    k_gemm16<<<1536, 256, 0, stream>>>(xh, wh, qkvf, 256, 48, Dz, E3z);
    k_gates<<<8192, 256, 0, stream>>>(x, agw, vgw, ga, gv);
    k_post<<<8192, 256, 0, stream>>>(qkvf, ve, cosp, sinp, gv, qh, kh, vT);
    k_attn<<<512, 256, 0, stream>>>(qh, kh, vT, ga, docs, scale, yh);
    // out = yh . wh[3D:4D]^T : M=4096, N=1024 (16 tiles of 64), K=1024
    k_gemm16<<<512, 256, 0, stream>>>(yh, wh + (size_t)E3z * Dz, out, 256, 16, Dz, Dz);
}
